// RNNTSP_4406636445697
// MI455X (gfx1250) — matrix operand order repaired and checked by reading, not compiled, not run on hardware
//
#include <hip/hip_runtime.h>
#include <math.h>

#define Bn   2048
#define Sn   100
#define En   256
#define Hn   256
#define G4H  (4*Hn)
#define Kc   256        // contraction dim: every GEMM in this model has K=256

typedef __attribute__((ext_vector_type(16))) _Float16 v16h;
typedef __attribute__((ext_vector_type(8)))  float    v8f;

union Frag16 { v16h h; unsigned u[8]; };

// ---------------------------------------------------------------------------
// Register-blocked WMMA GEMM (K hardcoded = 256, fully unrolled / pipelined):
//   C[M,N] (+)= A[M,256] @ W[N,256]^T (+bias[N])
// A, W are f16 row-major. One wave computes a 2x4 block of 16x16 C tiles
// (32 rows x 64 cols): 2 A-frags + 4 B-frags -> 8 v_wmma per 32-wide K chunk,
// 64 v_wmma total, with all fragment loads hoistable by the scheduler.
// Fragment layouts per CDNA5 ISA 7.12.2 (wave32):
//   A 16x32 f16 : lane r=l&15 (row), half=l>>4 ;  V[v] holds K = (v&3)*2 + (v>>2)*16 + half*8 (+1)
//   B 32x16 f16 : lane r=l&15 (col), half=l>>4 ;  V[v] holds K = (v&3)*2 + (v>>2)*16 + half*8 (+1)
//   C/D 16x16   : lane n = l&15, row m = v + (l>>4)*8
// Requires: M % 32 == 0, N % 64 == 0 (true for all uses here).
// ---------------------------------------------------------------------------
__global__ void gemm_f16_wmma(const _Float16* __restrict__ A, int lda,
                              const _Float16* __restrict__ W, int ldw,
                              void* __restrict__ Cp, int ldc,
                              int M, int N,
                              const float* __restrict__ bias,
                              int addC, int outHalf)
{
    const int wavesPerBlock = blockDim.x >> 5;
    int wave   = blockIdx.x * wavesPerBlock + (threadIdx.x >> 5);
    int nBlkN  = N >> 6;                       // 64-wide column blocks
    int totBlk = (M >> 5) * nBlkN;             // 32-tall row blocks
    if (wave >= totBlk) return;                // wave-uniform exit (EXEC stays all-1s)

    int bM = wave / nBlkN, bN = wave % nBlkN;
    int m0 = bM << 5, n0 = bN << 6;
    int lane = threadIdx.x & 31;
    int r    = lane & 15;
    int half = lane >> 4;

    v8f acc[2][4];
#pragma unroll
    for (int i = 0; i < 2; ++i)
#pragma unroll
        for (int j = 0; j < 4; ++j) acc[i][j] = (v8f){};

    if (addC) {
        const float* C = (const float*)Cp;
#pragma unroll
        for (int i = 0; i < 2; ++i)
#pragma unroll
            for (int j = 0; j < 4; ++j)
#pragma unroll
                for (int v = 0; v < 8; ++v)
                    acc[i][j][v] = C[(size_t)(m0 + i * 16 + v + half * 8) * ldc
                                     + n0 + j * 16 + r];
    }

    const _Float16* arow0 = A + (size_t)(m0 + r) * lda;
    const _Float16* arow1 = A + (size_t)(m0 + 16 + r) * lda;
    const _Float16* brow0 = W + (size_t)(n0 + r) * ldw;

#pragma unroll
    for (int k0 = 0; k0 < Kc; k0 += 32) {
        Frag16 fa[2], fb[4];
#pragma unroll
        for (int v = 0; v < 8; ++v) {
            int ka = k0 + ((v & 3) << 1) + ((v >> 2) << 4) + (half << 3);
            fa[0].u[v] = *(const unsigned*)(arow0 + ka);
            fa[1].u[v] = *(const unsigned*)(arow1 + ka);
        }
#pragma unroll
        for (int j = 0; j < 4; ++j) {
            const _Float16* br = brow0 + (size_t)j * 16 * ldw;
#pragma unroll
            for (int v = 0; v < 8; ++v) {
                int kb = k0 + ((v & 3) << 1) + ((v >> 2) << 4) + (half << 3);
                fb[j].u[v] = *(const unsigned*)(br + kb);
            }
        }
#pragma unroll
        for (int i = 0; i < 2; ++i)
#pragma unroll
            for (int j = 0; j < 4; ++j)
                acc[i][j] = __builtin_amdgcn_wmma_f32_16x16x32_f16(
                                false, fa[i].h, false, fb[j].h,
                                (short)0, acc[i][j], false, false);
    }

#pragma unroll
    for (int i = 0; i < 2; ++i)
#pragma unroll
        for (int j = 0; j < 4; ++j) {
            float bv = bias ? bias[n0 + j * 16 + r] : 0.0f;
#pragma unroll
            for (int v = 0; v < 8; ++v) {
                float  val = acc[i][j][v] + bv;
                size_t off = (size_t)(m0 + i * 16 + v + half * 8) * ldc
                           + n0 + j * 16 + r;
                if (outHalf) ((_Float16*)Cp)[off] = (_Float16)val;
                else         ((float*)Cp)[off]    = val;
            }
        }
}

// ---------------------------------------------------------------------------
// Elementwise / small kernels
// ---------------------------------------------------------------------------
__global__ void f32_to_f16_k(const float* __restrict__ s, _Float16* __restrict__ d, int n)
{
    int i = blockIdx.x * blockDim.x + threadIdx.x;
    if (i < n) d[i] = (_Float16)s[i];
}

__global__ void add_bias_k(const float* __restrict__ a, const float* __restrict__ b,
                           float* __restrict__ o, int n)
{
    int i = blockIdx.x * blockDim.x + threadIdx.x;
    if (i < n) o[i] = a[i] + b[i];
}

__global__ void zero_u32_k(unsigned* __restrict__ p, int n)
{
    int i = blockIdx.x * blockDim.x + threadIdx.x;
    if (i < n) p[i] = 0u;
}

// embedded[b,s,e] = x[b,s,0]*W[e,0] + x[b,s,1]*W[e,1] + bias[e]  (stored f16)
__global__ void embed_k(const float* __restrict__ x, const float* __restrict__ Wt,
                        const float* __restrict__ bias, _Float16* __restrict__ out)
{
    size_t idx = (size_t)blockIdx.x * blockDim.x + threadIdx.x;
    if (idx >= (size_t)Bn * Sn * En) return;
    size_t r = idx / En; int e = (int)(idx % En);
    float v = x[r * 2] * Wt[e * 2] + x[r * 2 + 1] * Wt[e * 2 + 1] + bias[e];
    out[idx] = (_Float16)v;
}

// LSTM cell: gates [B,4H] (i,f,g,o) f32  ->  c (f32, in/out), h (f16), optional enc_out slot
__global__ void lstm_cell_k(const float* __restrict__ gates, float* __restrict__ c,
                            _Float16* __restrict__ hh, _Float16* __restrict__ enc_slot)
{
    int idx = blockIdx.x * blockDim.x + threadIdx.x;
    if (idx >= Bn * Hn) return;
    int b = idx / Hn, j = idx % Hn;
    const float* g = gates + (size_t)b * G4H;
    float ig = 1.0f / (1.0f + expf(-g[j]));
    float fg = 1.0f / (1.0f + expf(-g[Hn + j]));
    float gg = tanhf(g[2 * Hn + j]);
    float og = 1.0f / (1.0f + expf(-g[3 * Hn + j]));
    float cn = fg * c[idx] + ig * gg;
    float hn = og * tanhf(cn);
    c[idx]  = cn;
    hh[idx] = (_Float16)hn;
    if (enc_slot) enc_slot[(size_t)b * Sn * Hn + j] = (_Float16)hn;
}

// logits[b,s] = mask ? -1e30 : Cclip * tanh( sum_h tanh(q[b,h]+k[b,s,h]) * V[h] )
// one wave per (b,s); each lane owns a contiguous 8-wide h-slice -> b128 loads
__global__ void attn_logits_k(const float* __restrict__ q, const _Float16* __restrict__ k,
                              const float* __restrict__ V, const float* __restrict__ mask,
                              float* __restrict__ logits, float Cclip)
{
    int gw = (int)((blockIdx.x * (size_t)blockDim.x + threadIdx.x) >> 5);
    int lane = threadIdx.x & 31;
    if (gw >= Bn * Sn) return;
    int b = gw / Sn;
    int h0 = lane * 8;                           // Hn = 256 = 32 lanes * 8

    union { uint4 u; _Float16 h[8]; } kv;
    kv.u = *(const uint4*)(k + (size_t)gw * Hn + h0);
    const float4* qp = (const float4*)(q + (size_t)b * Hn + h0);
    const float4* vp = (const float4*)(V + h0);
    float4 q0 = qp[0], q1 = qp[1];
    float4 v0 = vp[0], v1 = vp[1];

    float sum = 0.0f;
    sum += tanhf(q0.x + (float)kv.h[0]) * v0.x;
    sum += tanhf(q0.y + (float)kv.h[1]) * v0.y;
    sum += tanhf(q0.z + (float)kv.h[2]) * v0.z;
    sum += tanhf(q0.w + (float)kv.h[3]) * v0.w;
    sum += tanhf(q1.x + (float)kv.h[4]) * v1.x;
    sum += tanhf(q1.y + (float)kv.h[5]) * v1.y;
    sum += tanhf(q1.z + (float)kv.h[6]) * v1.z;
    sum += tanhf(q1.w + (float)kv.h[7]) * v1.w;

#pragma unroll
    for (int off = 16; off; off >>= 1) sum += __shfl_xor(sum, off, 32);
    if (lane == 0) {
        float lg = Cclip * tanhf(sum);
        logits[gw] = (mask[gw] > 0.5f) ? -1e30f : lg;
    }
}

// query[b,:] = sum_s softmax(logits[b,:])[s] * k_gl[b,s,:]   (one block of 256 = H per b)
__global__ void softmax_glimpse_k(const float* __restrict__ logits,
                                  const _Float16* __restrict__ kgl,
                                  _Float16* __restrict__ query)
{
    __shared__ float w[Sn];
    __shared__ float inv;
    int b = blockIdx.x, tid = threadIdx.x;
    if (tid < Sn) w[tid] = logits[(size_t)b * Sn + tid];
    __syncthreads();
    if (tid == 0) {
        float mx = -1e30f;
        for (int s = 0; s < Sn; ++s) mx = fmaxf(mx, w[s]);
        float sum = 0.0f;
        for (int s = 0; s < Sn; ++s) { float e = expf(w[s] - mx); w[s] = e; sum += e; }
        inv = 1.0f / sum;
    }
    __syncthreads();
    float acc = 0.0f;
    const _Float16* kb = kgl + (size_t)b * Sn * Hn + tid;
    for (int s = 0; s < Sn; ++s) acc += w[s] * (float)kb[(size_t)s * Hn];
    query[(size_t)b * Hn + tid] = (_Float16)(acc * inv);
}

__device__ inline float u01hash(unsigned long long x)
{
    x ^= x >> 33; x *= 0xff51afd7ed558ccdULL;
    x ^= x >> 33; x *= 0xc4ceb9fe1a85ec53ULL;
    x ^= x >> 33;
    return ((float)(x >> 40) + 0.5f) * (1.0f / 16777216.0f);
}

// Gumbel-argmax categorical sample + log-prob; one thread per b
__global__ void sample_step_k(const float* __restrict__ logits, int step,
                              float* __restrict__ logp_out, float* __restrict__ idx_out,
                              int* __restrict__ chosen, float* __restrict__ mask)
{
    int b = blockIdx.x * blockDim.x + threadIdx.x;
    if (b >= Bn) return;
    const float* l = logits + (size_t)b * Sn;
    float mx = -1e30f;
    for (int s = 0; s < Sn; ++s) mx = fmaxf(mx, l[s]);
    float sum = 0.0f;
    for (int s = 0; s < Sn; ++s) sum += expf(l[s] - mx);
    int best = 0; float bestv = -3e38f;
    for (int s = 0; s < Sn; ++s) {
        unsigned long long key = (((unsigned long long)(step * Bn + b)) << 32)
                               ^ (unsigned long long)((unsigned)s * 2654435761u)
                               ^ 0x9E3779B97F4A7C15ULL;
        float u = u01hash(key);
        float v = l[s] + (-logf(-logf(u)));
        if (v > bestv) { bestv = v; best = s; }
    }
    float logp = l[best] - mx - logf(sum);
    logp_out[(size_t)b * Sn + step] = logp;
    idx_out [(size_t)b * Sn + step] = (float)best;
    chosen[b] = best;
    mask[(size_t)b * Sn + best] = 1.0f;
}

__global__ void gather_decin_k(const _Float16* __restrict__ emb, const int* __restrict__ chosen,
                               _Float16* __restrict__ dec_in)
{
    int idx = blockIdx.x * blockDim.x + threadIdx.x;
    if (idx >= Bn * En) return;
    int b = idx / En, e = idx % En;
    dec_in[idx] = emb[((size_t)b * Sn + chosen[b]) * (size_t)En + e];
}

__global__ void init_decin_k(const float* __restrict__ sos, _Float16* __restrict__ dec_in)
{
    int idx = blockIdx.x * blockDim.x + threadIdx.x;
    if (idx >= Bn * En) return;
    dec_in[idx] = (_Float16)sos[idx % En];
}

// ---------------------------------------------------------------------------
// Host orchestration
// ---------------------------------------------------------------------------
static inline int cdiv_i(long long a, long long b) { return (int)((a + b - 1) / b); }

static void gemm(hipStream_t st, const void* A, int lda, const void* W, int ldw,
                 void* C, int ldc, int M, int N,
                 const void* bias, int addC, int outHalf)
{
    int blkTiles = (M / 32) * (N / 64);        // one wave per 32x64 block
    int blocks   = cdiv_i(blkTiles, 4);
    gemm_f16_wmma<<<blocks, 128, 0, st>>>((const _Float16*)A, lda, (const _Float16*)W, ldw,
                                          C, ldc, M, N, (const float*)bias, addC, outHalf);
}

extern "C" void kernel_launch(void* const* d_in, const int* in_sizes, int n_in,
                              void* d_out, int out_size, void* d_ws, size_t ws_size,
                              hipStream_t stream)
{
    (void)in_sizes; (void)n_in; (void)out_size; (void)ws_size;
    const float* inputs  = (const float*)d_in[0];
    const float* emb_W   = (const float*)d_in[1];
    const float* emb_b   = (const float*)d_in[2];
    const float* enc_Wih = (const float*)d_in[3];
    const float* enc_Whh = (const float*)d_in[4];
    const float* enc_bih = (const float*)d_in[5];
    const float* enc_bhh = (const float*)d_in[6];
    const float* dec_Wih = (const float*)d_in[7];
    const float* dec_Whh = (const float*)d_in[8];
    const float* dec_bih = (const float*)d_in[9];
    const float* dec_bhh = (const float*)d_in[10];
    const float* ptr_Wq  = (const float*)d_in[11];
    const float* ptr_bq  = (const float*)d_in[12];
    const float* ptr_Wk  = (const float*)d_in[13];
    const float* ptr_bk  = (const float*)d_in[14];
    const float* ptr_V   = (const float*)d_in[15];
    const float* gl_Wq   = (const float*)d_in[16];
    const float* gl_bq   = (const float*)d_in[17];
    const float* gl_Wk   = (const float*)d_in[18];
    const float* gl_bk   = (const float*)d_in[19];
    const float* gl_V    = (const float*)d_in[20];
    const float* dec_sos = (const float*)d_in[21];

    // ---- workspace layout ----
    char* base = (char*)d_ws; size_t off = 0;
    auto alloc = [&](size_t bytes) -> void* {
        void* p = base + off; off = (off + bytes + 255) & ~(size_t)255; return p;
    };
    _Float16* emb_h     = (_Float16*)alloc((size_t)Bn * Sn * En * 2);
    _Float16* enc_out_h = (_Float16*)alloc((size_t)Bn * Sn * Hn * 2);
    _Float16* k_ptr_h   = (_Float16*)alloc((size_t)Bn * Sn * Hn * 2);
    _Float16* k_gl_h    = (_Float16*)alloc((size_t)Bn * Sn * Hn * 2);
    float*    gates     = (float*)   alloc((size_t)Bn * G4H * 4);
    float*    c_f32     = (float*)   alloc((size_t)Bn * Hn * 4);
    _Float16* h_h       = (_Float16*)alloc((size_t)Bn * Hn * 2);
    float*    qbuf      = (float*)   alloc((size_t)Bn * Hn * 4);
    _Float16* query_h   = (_Float16*)alloc((size_t)Bn * Hn * 2);
    _Float16* dec_in_h  = (_Float16*)alloc((size_t)Bn * En * 2);
    float*    logits    = (float*)   alloc((size_t)Bn * Sn * 4);
    float*    maskb     = (float*)   alloc((size_t)Bn * Sn * 4);
    int*      chosen    = (int*)     alloc((size_t)Bn * 4);
    float*    enc_bsum  = (float*)   alloc((size_t)G4H * 4);
    float*    dec_bsum  = (float*)   alloc((size_t)G4H * 4);
    _Float16* enc_Wih_h = (_Float16*)alloc((size_t)G4H * En * 2);
    _Float16* enc_Whh_h = (_Float16*)alloc((size_t)G4H * Hn * 2);
    _Float16* dec_Wih_h = (_Float16*)alloc((size_t)G4H * En * 2);
    _Float16* dec_Whh_h = (_Float16*)alloc((size_t)G4H * Hn * 2);
    _Float16* ptr_Wq_h  = (_Float16*)alloc((size_t)Hn * Hn * 2);
    _Float16* ptr_Wk_h  = (_Float16*)alloc((size_t)Hn * Hn * 2);
    _Float16* gl_Wq_h   = (_Float16*)alloc((size_t)Hn * Hn * 2);
    _Float16* gl_Wk_h   = (_Float16*)alloc((size_t)Hn * Hn * 2);

    float* out_logp = (float*)d_out;
    float* out_idx  = (float*)d_out + (size_t)Bn * Sn;

    const int T = 256;

    // ---- weight conversion (f32 -> f16) + fused biases ----
    f32_to_f16_k<<<cdiv_i(G4H * En, T), T, 0, stream>>>(enc_Wih, enc_Wih_h, G4H * En);
    f32_to_f16_k<<<cdiv_i(G4H * Hn, T), T, 0, stream>>>(enc_Whh, enc_Whh_h, G4H * Hn);
    f32_to_f16_k<<<cdiv_i(G4H * En, T), T, 0, stream>>>(dec_Wih, dec_Wih_h, G4H * En);
    f32_to_f16_k<<<cdiv_i(G4H * Hn, T), T, 0, stream>>>(dec_Whh, dec_Whh_h, G4H * Hn);
    f32_to_f16_k<<<cdiv_i(Hn * Hn, T), T, 0, stream>>>(ptr_Wq, ptr_Wq_h, Hn * Hn);
    f32_to_f16_k<<<cdiv_i(Hn * Hn, T), T, 0, stream>>>(ptr_Wk, ptr_Wk_h, Hn * Hn);
    f32_to_f16_k<<<cdiv_i(Hn * Hn, T), T, 0, stream>>>(gl_Wq,  gl_Wq_h,  Hn * Hn);
    f32_to_f16_k<<<cdiv_i(Hn * Hn, T), T, 0, stream>>>(gl_Wk,  gl_Wk_h,  Hn * Hn);
    add_bias_k<<<cdiv_i(G4H, T), T, 0, stream>>>(enc_bih, enc_bhh, enc_bsum, G4H);
    add_bias_k<<<cdiv_i(G4H, T), T, 0, stream>>>(dec_bih, dec_bhh, dec_bsum, G4H);

    // ---- embedding ----
    embed_k<<<cdiv_i((size_t)Bn * Sn * En, T), T, 0, stream>>>(inputs, emb_W, emb_b, emb_h);

    // ---- state init (deterministic per-call) ----
    zero_u32_k<<<cdiv_i(Bn * Hn, T), T, 0, stream>>>((unsigned*)c_f32, Bn * Hn);
    zero_u32_k<<<cdiv_i(Bn * Hn / 2, T), T, 0, stream>>>((unsigned*)h_h, Bn * Hn / 2);
    zero_u32_k<<<cdiv_i(Bn * Sn, T), T, 0, stream>>>((unsigned*)maskb, Bn * Sn);

    // ---- encoder LSTM (100 sequential steps, two WMMA GEMMs + cell each) ----
    for (int t = 0; t < Sn; ++t) {
        gemm(stream, emb_h + (size_t)t * En, Sn * En, enc_Wih_h, En,
             gates, G4H, Bn, G4H, enc_bsum, 0, 0);
        gemm(stream, h_h, Hn, enc_Whh_h, Hn,
             gates, G4H, Bn, G4H, nullptr, 1, 0);
        lstm_cell_k<<<cdiv_i(Bn * Hn, T), T, 0, stream>>>(gates, c_f32, h_h,
                                                          enc_out_h + (size_t)t * Hn);
    }

    // ---- key projections (big WMMA GEMMs over B*S rows, f16 output) ----
    gemm(stream, enc_out_h, Hn, ptr_Wk_h, Hn, k_ptr_h, Hn, Bn * Sn, Hn, ptr_bk, 0, 1);
    gemm(stream, enc_out_h, Hn, gl_Wk_h,  Hn, k_gl_h,  Hn, Bn * Sn, Hn, gl_bk,  0, 1);

    // ---- decoder ----
    init_decin_k<<<cdiv_i(Bn * En, T), T, 0, stream>>>(dec_sos, dec_in_h);
    for (int step = 0; step < Sn; ++step) {
        gemm(stream, dec_in_h, En, dec_Wih_h, En, gates, G4H, Bn, G4H, dec_bsum, 0, 0);
        gemm(stream, h_h, Hn, dec_Whh_h, Hn, gates, G4H, Bn, G4H, nullptr, 1, 0);
        lstm_cell_k<<<cdiv_i(Bn * Hn, T), T, 0, stream>>>(gates, c_f32, h_h, nullptr);

        // glimpse attention
        gemm(stream, h_h, Hn, gl_Wq_h, Hn, qbuf, Hn, Bn, Hn, gl_bq, 0, 0);
        attn_logits_k<<<cdiv_i((long long)Bn * Sn * 32, T), T, 0, stream>>>(
            qbuf, k_gl_h, gl_V, maskb, logits, 10.0f);
        softmax_glimpse_k<<<Bn, Hn, 0, stream>>>(logits, k_gl_h, query_h);

        // pointer attention
        gemm(stream, query_h, Hn, ptr_Wq_h, Hn, qbuf, Hn, Bn, Hn, ptr_bq, 0, 0);
        attn_logits_k<<<cdiv_i((long long)Bn * Sn * 32, T), T, 0, stream>>>(
            qbuf, k_ptr_h, ptr_V, maskb, logits, 10.0f);

        // sample, log-prob, mask update, next input
        sample_step_k<<<cdiv_i(Bn, T), T, 0, stream>>>(logits, step, out_logp, out_idx,
                                                       chosen, maskb);
        gather_decin_k<<<cdiv_i(Bn * En, T), T, 0, stream>>>(emb_h, chosen, dec_in_h);
    }
}
